// RNNSeq2Seq_79800492359723
// MI455X (gfx1250) — matrix operand order repaired and checked by reading, not compiled, not run on hardware
//
#include <hip/hip_runtime.h>

#define BB 512
#define TT 24
#define TE 23
#define DD 1024
#define D3 3072
#define NOUT 325
#define HZ 12
#define TSLOT 288
#define KPE 352    // K=325 padded to mult of 32 (embedding GEMMs)
#define NPM 384    // N=325 padded to mult of 128 (MLP head GEMM)
#define ME (BB*TE) // 11776 encoder rows

typedef __attribute__((ext_vector_type(16))) _Float16 v16h;
typedef __attribute__((ext_vector_type(8)))  _Float16 v8h;
typedef __attribute__((ext_vector_type(8)))  float    v8f;

union H16 { struct { v8h lo; v8h hi; } p; v16h v; };

// A fragment (16x32 MxK, f16): lane holds 16 contiguous halves (+0, +8)
__device__ inline v16h ldA16(const _Float16* __restrict__ p) {
  H16 u;
  u.p.lo = *(const v8h*)(p);
  u.p.hi = *(const v8h*)(p + 8);
  return u.v;
}
// B fragment (32x16 KxN, f16): lane holds 16 contiguous halves
__device__ inline v16h ldB16(const _Float16* __restrict__ p) {
  return *(const v16h*)p;
}

#define WMMA(A_, B_, C_) __builtin_amdgcn_wmma_f32_16x16x32_f16(false, A_, false, B_, (short)0, C_, false, false)

// ---------------------------------------------------------------------------
// WMMA GEMM: C[M,N] = A[M,K](f16) * W[N,K](f16)^T + bias[N], f32 accumulate.
// block = 128 threads = 4 waves; block tile 64x128; wave tile 32x64.
// Per k-step: 12 x b128 loads, 8 WMMAs (1.5 loads/WMMA), register double-buffer.
// Requires M%64==0, N%128==0, K%32==0 (guaranteed by padding).
// ---------------------------------------------------------------------------
__global__ __launch_bounds__(128)
void k_gemm_wmma(const _Float16* __restrict__ A, int lda,
                 const _Float16* __restrict__ W, int ldk,
                 const float* __restrict__ bias,
                 float* __restrict__ C, int ldc, int ksteps)
{
  const int lane = threadIdx.x & 31;
  const int wv   = threadIdx.x >> 5;
  const int mBase = blockIdx.y * 64  + ((wv >> 1) << 5);
  const int nBase = blockIdx.x * 128 + ((wv & 1) << 6);

  const int l15 = lane & 15;
  const int hi  = lane >> 4;

  v8f acc[2][4];
#pragma unroll
  for (int i = 0; i < 2; ++i)
#pragma unroll
    for (int j = 0; j < 4; ++j)
      acc[i][j] = (v8f){0.f,0.f,0.f,0.f,0.f,0.f,0.f,0.f};

  // A: lane l -> row mBase+l%16, halves j=0..15 <- K=kb+hi*16+j contiguous
  const _Float16* pA0 = A + (size_t)(mBase + l15) * lda + (hi << 4);
  const _Float16* pA1 = pA0 + (size_t)16 * lda;
  // B: lane l -> col nBase+16*j+l%16, halves 0..15 <- K=kb+hi*16.. contiguous
  const _Float16* pB0 = W + (size_t)(nBase + l15) * ldk + (hi << 4);
  const _Float16* pB1 = pB0 + (size_t)16 * ldk;
  const _Float16* pB2 = pB0 + (size_t)32 * ldk;
  const _Float16* pB3 = pB0 + (size_t)48 * ldk;

  v16h a0 = ldA16(pA0), a1 = ldA16(pA1);
  v16h b0 = ldB16(pB0), b1 = ldB16(pB1), b2 = ldB16(pB2), b3 = ldB16(pB3);

  for (int ks = 0; ks + 1 < ksteps; ++ks) {
    const int kn = (ks + 1) << 5;
    // issue next-iteration loads first (overlap with WMMAs below)
    v16h a0n = ldA16(pA0 + kn);
    v16h a1n = ldA16(pA1 + kn);
    v16h b0n = ldB16(pB0 + kn);
    v16h b1n = ldB16(pB1 + kn);
    v16h b2n = ldB16(pB2 + kn);
    v16h b3n = ldB16(pB3 + kn);
    acc[0][0] = WMMA(a0, b0, acc[0][0]);
    acc[1][0] = WMMA(a1, b0, acc[1][0]);
    acc[0][1] = WMMA(a0, b1, acc[0][1]);
    acc[1][1] = WMMA(a1, b1, acc[1][1]);
    acc[0][2] = WMMA(a0, b2, acc[0][2]);
    acc[1][2] = WMMA(a1, b2, acc[1][2]);
    acc[0][3] = WMMA(a0, b3, acc[0][3]);
    acc[1][3] = WMMA(a1, b3, acc[1][3]);
    a0 = a0n; a1 = a1n; b0 = b0n; b1 = b1n; b2 = b2n; b3 = b3n;
  }
  acc[0][0] = WMMA(a0, b0, acc[0][0]);
  acc[1][0] = WMMA(a1, b0, acc[1][0]);
  acc[0][1] = WMMA(a0, b1, acc[0][1]);
  acc[1][1] = WMMA(a1, b1, acc[1][1]);
  acc[0][2] = WMMA(a0, b2, acc[0][2]);
  acc[1][2] = WMMA(a1, b2, acc[1][2]);
  acc[0][3] = WMMA(a0, b3, acc[0][3]);
  acc[1][3] = WMMA(a1, b3, acc[1][3]);

  // C/D layout: lane l -> col N = base + l%16 ; vgpr r -> row M = base + r + 8*(l/16)
  const int rsel = hi << 3;
#pragma unroll
  for (int j = 0; j < 4; ++j) {
    const int col = nBase + (j << 4) + l15;
    const float bv = bias[col];
#pragma unroll
    for (int r = 0; r < 8; ++r) {
      const int row0 = mBase + rsel + r;
      C[(size_t)row0 * ldc + col]        = acc[0][j][r] + bv;
      C[(size_t)(row0 + 16) * ldc + col] = acc[1][j][r] + bv;
    }
  }
}

// ---------------------------------------------------------------------------
// Support kernels
// ---------------------------------------------------------------------------
__global__ void k_cvt_f16(const float* __restrict__ s, _Float16* __restrict__ d, int n) {
  int i = blockIdx.x * blockDim.x + threadIdx.x;
  if (i < n) d[i] = (_Float16)s[i];
}

__global__ void k_zero2(float* __restrict__ p32, _Float16* __restrict__ p16, int n) {
  int i = blockIdx.x * blockDim.x + threadIdx.x;
  if (i < n) { p32[i] = 0.f; p16[i] = (_Float16)0.f; }
}

// w_data (325,1024) -> wdT (1024 rows, KPE) f16, zero-padded K
__global__ void k_wdata_T(const float* __restrict__ w, _Float16* __restrict__ d) {
  int i = blockIdx.x * blockDim.x + threadIdx.x;
  if (i >= DD * KPE) return;
  int n = i / KPE, k = i % KPE;
  d[i] = (k < NOUT) ? (_Float16)w[(size_t)k * DD + n] : (_Float16)0.f;
}

// w_mlp (1024,325) -> wmT (NPM rows, 1024) f16, zero-padded N rows
__global__ void k_wmlp_T(const float* __restrict__ w, _Float16* __restrict__ d) {
  int i = blockIdx.x * blockDim.x + threadIdx.x;
  if (i >= NPM * DD) return;
  int n = i / DD, k = i % DD;
  d[i] = (n < NOUT) ? (_Float16)w[(size_t)k * NOUT + n] : (_Float16)0.f;
}

__global__ void k_pad_bias(const float* __restrict__ b, float* __restrict__ d) {
  int i = blockIdx.x * blockDim.x + threadIdx.x;
  if (i < NPM) d[i] = (i < NOUT) ? b[i] : 0.f;
}

// data[:, :-1] -> x0p (ME, KPE) f16, zero-padded
__global__ void k_pad_data_enc(const float* __restrict__ data, _Float16* __restrict__ x0p) {
  int i = blockIdx.x * blockDim.x + threadIdx.x;
  if (i >= ME * KPE) return;
  int r = i / KPE, k = i % KPE;
  int b = r / TE, t = r % TE;
  x0p[i] = (k < NOUT) ? (_Float16)data[((size_t)b * TT + t) * NOUT + k] : (_Float16)0.f;
}

// decoder init: data[:, -1] -> xdP (BB, KPE) f16 padded and curD (BB, NOUT) f32
__global__ void k_init_dec(const float* __restrict__ data, _Float16* __restrict__ xdP,
                           float* __restrict__ curD) {
  int i = blockIdx.x * blockDim.x + threadIdx.x;
  if (i >= BB * KPE) return;
  int b = i / KPE, k = i % KPE;
  float v = (k < NOUT) ? data[((size_t)b * TT + (TT - 1)) * NOUT + k] : 0.f;
  xdP[i] = (_Float16)v;
  if (k < NOUT) curD[b * NOUT + k] = v;
}

// xeh = f16(xe + time_table[time[b,t] % 288] + day_table[weekday[b]])
__global__ void k_tables_enc(const float* __restrict__ xe, _Float16* __restrict__ xeh,
                             const int* __restrict__ tm, const int* __restrict__ wd,
                             const float* __restrict__ tt, const float* __restrict__ dt) {
  int i = blockIdx.x * blockDim.x + threadIdx.x;
  if (i >= ME * DD) return;
  int d = i & (DD - 1);
  int r = i >> 10;
  int b = r / TE, t = r % TE;
  int ts = tm[b * TT + t] % TSLOT; if (ts < 0) ts += TSLOT;
  xeh[i] = (_Float16)(xe[i] + tt[(size_t)ts * DD + d] + dt[(size_t)wd[b] * DD + d]);
}

__global__ void k_tables_dec(const float* __restrict__ xde, _Float16* __restrict__ xdeh,
                             const int* __restrict__ tm, const int* __restrict__ wd,
                             const float* __restrict__ tt, const float* __restrict__ dt,
                             int step) {
  int i = blockIdx.x * blockDim.x + threadIdx.x;
  if (i >= BB * DD) return;
  int d = i & (DD - 1);
  int b = i >> 10;
  int ts = (tm[b * TT + (TT - 1)] + step) % TSLOT; if (ts < 0) ts += TSLOT;
  xdeh[i] = (_Float16)(xde[i] + tt[(size_t)ts * DD + d] + dt[(size_t)wd[b] * DD + d]);
}

// GRU gate combine: h = (1-z)*n + z*h (f32 master + f16 mirror, optional ys f16)
__global__ void k_gru_combine(const float* __restrict__ gx, int gxMul, int gxAdd,
                              const float* __restrict__ gh, float* __restrict__ h,
                              _Float16* __restrict__ hh, _Float16* __restrict__ ys,
                              int ysAdd, int useYs) {
  int i = blockIdx.x * blockDim.x + threadIdx.x;
  if (i >= BB * DD) return;
  int d = i & (DD - 1);
  int b = i >> 10;
  const float* gxp = gx + (size_t)(b * gxMul + gxAdd) * D3;
  const float* ghp = gh + (size_t)b * D3;
  float r = 1.f / (1.f + __expf(-(gxp[d] + ghp[d])));
  float z = 1.f / (1.f + __expf(-(gxp[d + DD] + ghp[d + DD])));
  float n = tanhf(gxp[d + 2 * DD] + r * ghp[d + 2 * DD]);
  float hv = h[i];
  float hn = (1.f - z) * n + z * hv;
  h[i] = hn;
  hh[i] = (_Float16)hn;
  if (useYs) ys[(size_t)(b * TE + ysAdd) * DD + d] = (_Float16)hn;
}

// decoder output + feedback: out = curD + res ; curD = out ; xdP = f16(out)
__global__ void k_out_step(const float* __restrict__ resP, float* __restrict__ curD,
                           _Float16* __restrict__ xdP, float* __restrict__ out, int step) {
  int i = blockIdx.x * blockDim.x + threadIdx.x;
  if (i >= BB * NOUT) return;
  int b = i / NOUT, o = i % NOUT;
  float v = curD[i] + resP[(size_t)b * NPM + o];
  out[((size_t)b * HZ + step) * NOUT + o] = v;
  curD[i] = v;
  xdP[(size_t)b * KPE + o] = (_Float16)v;
}

// ---------------------------------------------------------------------------
extern "C" void kernel_launch(void* const* d_in, const int* in_sizes, int n_in,
                              void* d_out, int out_size, void* d_ws, size_t ws_size,
                              hipStream_t stream) {
  const float* data     = (const float*)d_in[0];
  const int*   tm       = (const int*)d_in[1];
  const int*   wd       = (const int*)d_in[2];
  const float* w_data   = (const float*)d_in[3];
  const float* b_data   = (const float*)d_in[4];
  const float* time_tab = (const float*)d_in[5];
  const float* day_tab  = (const float*)d_in[6];
  const float* enc_Wih  = (const float*)d_in[7];
  const float* enc_Whh  = (const float*)d_in[8];
  const float* enc_bih  = (const float*)d_in[9];
  const float* enc_bhh  = (const float*)d_in[10];
  const float* dec_Wih  = (const float*)d_in[11];
  const float* dec_Whh  = (const float*)d_in[12];
  const float* dec_bih  = (const float*)d_in[13];
  const float* dec_bhh  = (const float*)d_in[14];
  const float* w_mlp    = (const float*)d_in[15];
  const float* b_mlp    = (const float*)d_in[16];
  float* out = (float*)d_out;

  const size_t WELE = 2ULL * D3 * DD; // per weight family (L=2 layers)

  char* ws = (char*)d_ws;
  size_t off = 0;
  auto take = [&](size_t bytes) -> char* {
    char* p = ws + off;
    off = (off + bytes + 255) & ~(size_t)255;
    return p;
  };

  _Float16* encWihB = (_Float16*)take(WELE * 2);
  _Float16* encWhhB = (_Float16*)take(WELE * 2);
  _Float16* decWihB = (_Float16*)take(WELE * 2);
  _Float16* decWhhB = (_Float16*)take(WELE * 2);
  _Float16* wdT     = (_Float16*)take((size_t)DD * KPE * 2);
  _Float16* wmT     = (_Float16*)take((size_t)NPM * DD * 2);
  float*    bmlpP   = (float*)take((size_t)NPM * 4);
  _Float16* x0p     = (_Float16*)take((size_t)ME * KPE * 2);
  float*    xe      = (float*)take((size_t)ME * DD * 4);
  _Float16* xeh     = (_Float16*)take((size_t)ME * DD * 2);
  float*    gxAll   = (float*)take((size_t)ME * D3 * 4);
  float*    gh      = (float*)take((size_t)BB * D3 * 4);
  float*    hbuf    = (float*)take((size_t)2 * BB * DD * 4);
  _Float16* hbuf16  = (_Float16*)take((size_t)2 * BB * DD * 2);
  _Float16* xdP     = (_Float16*)take((size_t)BB * KPE * 2);
  float*    xde     = (float*)take((size_t)BB * DD * 4);
  _Float16* xdeh    = (_Float16*)take((size_t)BB * DD * 2);
  float*    gxd     = (float*)take((size_t)BB * D3 * 4);
  float*    resP    = (float*)take((size_t)BB * NPM * 4);
  float*    curD    = (float*)take((size_t)BB * NOUT * 4);
  if (off > ws_size) return; // workspace too small

  float*    h0  = hbuf;
  float*    h1  = hbuf + (size_t)BB * DD;
  _Float16* h0h = hbuf16;
  _Float16* h1h = hbuf16 + (size_t)BB * DD;

  auto ce = [](size_t n) { return (unsigned)((n + 255) / 256); };
  auto gemm = [&](const _Float16* A, int lda, const _Float16* W, int ldk,
                  const float* bias, float* C, int ldc, int M, int N, int K) {
    dim3 g((unsigned)(N / 128), (unsigned)(M / 64), 1);
    k_gemm_wmma<<<g, 128, 0, stream>>>(A, lda, W, ldk, bias, C, ldc, K / 32);
  };

  // --- weight conversion / padding (every launch; deterministic) ---
  k_cvt_f16<<<ce(WELE), 256, 0, stream>>>(enc_Wih, encWihB, (int)WELE);
  k_cvt_f16<<<ce(WELE), 256, 0, stream>>>(enc_Whh, encWhhB, (int)WELE);
  k_cvt_f16<<<ce(WELE), 256, 0, stream>>>(dec_Wih, decWihB, (int)WELE);
  k_cvt_f16<<<ce(WELE), 256, 0, stream>>>(dec_Whh, decWhhB, (int)WELE);
  k_wdata_T<<<ce((size_t)DD * KPE), 256, 0, stream>>>(w_data, wdT);
  k_wmlp_T<<<ce((size_t)NPM * DD), 256, 0, stream>>>(w_mlp, wmT);
  k_pad_bias<<<ce(NPM), 256, 0, stream>>>(b_mlp, bmlpP);

  // --- encoder embedding ---
  k_pad_data_enc<<<ce((size_t)ME * KPE), 256, 0, stream>>>(data, x0p);
  gemm(x0p, KPE, wdT, KPE, b_data, xe, DD, ME, DD, KPE);
  k_tables_enc<<<ce((size_t)ME * DD), 256, 0, stream>>>(xe, xeh, tm, wd, time_tab, day_tab);

  // --- encoder GRU (2 layers, 23 steps each) ---
  k_zero2<<<ce((size_t)2 * BB * DD), 256, 0, stream>>>(hbuf, hbuf16, 2 * BB * DD);
  for (int l = 0; l < 2; ++l) {
    const _Float16* Wih = encWihB + (size_t)l * D3 * DD;
    const _Float16* Whh = encWhhB + (size_t)l * D3 * DD;
    const float* bih = enc_bih + (size_t)l * D3;
    const float* bhh = enc_bhh + (size_t)l * D3;
    float*    h  = l ? h1 : h0;
    _Float16* hh = l ? h1h : h0h;
    // batched input-gate GEMM across all 23 timesteps
    gemm(xeh, DD, Wih, DD, bih, gxAll, D3, ME, D3, DD);
    for (int t = 0; t < TE; ++t) {
      gemm(hh, DD, Whh, DD, bhh, gh, D3, BB, D3, DD);
      // record layer output into xeh (feeds next layer's batched GEMM)
      k_gru_combine<<<ce((size_t)BB * DD), 256, 0, stream>>>(gxAll, TE, t, gh, h, hh, xeh, t, 1);
    }
  }

  // --- decoder (12 autoregressive steps) ---
  k_init_dec<<<ce((size_t)BB * KPE), 256, 0, stream>>>(data, xdP, curD);
  for (int s = 0; s < HZ; ++s) {
    gemm(xdP, KPE, wdT, KPE, b_data, xde, DD, BB, DD, KPE);
    k_tables_dec<<<ce((size_t)BB * DD), 256, 0, stream>>>(xde, xdeh, tm, wd, time_tab, day_tab, s);
    // layer 0
    gemm(xdeh, DD, decWihB, DD, dec_bih, gxd, D3, BB, D3, DD);
    gemm(h0h, DD, decWhhB, DD, dec_bhh, gh, D3, BB, D3, DD);
    k_gru_combine<<<ce((size_t)BB * DD), 256, 0, stream>>>(gxd, 1, 0, gh, h0, h0h, xeh, 0, 0);
    // layer 1
    gemm(h0h, DD, decWihB + (size_t)D3 * DD, DD, dec_bih + D3, gxd, D3, BB, D3, DD);
    gemm(h1h, DD, decWhhB + (size_t)D3 * DD, DD, dec_bhh + D3, gh, D3, BB, D3, DD);
    k_gru_combine<<<ce((size_t)BB * DD), 256, 0, stream>>>(gxd, 1, 0, gh, h1, h1h, xeh, 0, 0);
    // MLP head + output/feedback
    gemm(h1h, DD, wmT, DD, bmlpP, resP, NPM, BB, NPM, DD);
    k_out_step<<<ce((size_t)BB * NOUT), 256, 0, stream>>>(resP, curD, xdP, out, s);
  }
}
